// SimpleRNN_84370337562780
// MI455X (gfx1250) — compile-verified, matrix operand order checked by reading, not run on hardware
//
#include <hip/hip_runtime.h>
#include <hip/hip_bf16.h>

#define B_   64
#define T_   4096
#define V_   256
#define H_   32
#define OUT_ 4

typedef __attribute__((ext_vector_type(16))) _Float16     v16h;
typedef __attribute__((ext_vector_type(8)))  float        v8f;
typedef __attribute__((ext_vector_type(4)))  unsigned int u32x4;
typedef __attribute__((ext_vector_type(4)))  float        f32x4;

union V16H { v16h v; _Float16 h[16]; u32x4 q[2]; };

// ---------------------------------------------------------------------------
// A-fragment loader: 16x32 f16 tile, row-major source with pitch 32 elements.
// Layout per ISA 7.12.2: lanes 0-15 rows M=0..15 K=0..7 / 16..23,
//                        lanes 16-31 rows M=0..15 K=8..15 / 24..31.
// ---------------------------------------------------------------------------
__device__ __forceinline__ v16h load_a16x32(const _Float16* rowmaj, int lane) {
    int m  = lane & 15;
    int k0 = (lane & 16) ? 8 : 0;
    const _Float16* p = rowmaj + m * 32 + k0;
    V16H r;
    r.q[0] = *(const u32x4*)(p);        // K = k0 .. k0+7
    r.q[1] = *(const u32x4*)(p + 16);   // K = k0+16 .. k0+23
    return r.v;
}

// ---------------------------------------------------------------------------
// B-fragment loader from an f32 weight matrix W[N][K] (row-major, pitch 32):
// B[k][n] = W[n][k], i.e. computes X @ W^T. Column n striped like A's rows.
// ---------------------------------------------------------------------------
__device__ __forceinline__ v16h load_b_from_w32(const float* __restrict__ W,
                                                int n0, int lane) {
    int n  = n0 + (lane & 15);
    int k0 = (lane & 16) ? 8 : 0;
    const float* p = W + n * 32 + k0;
    V16H r;
#pragma unroll
    for (int i = 0; i < 8; ++i) {
        r.h[i]     = (_Float16)p[i];
        r.h[8 + i] = (_Float16)p[16 + i];
    }
    return r.v;
}

// ---------------------------------------------------------------------------
// Kernel 1: argmax over V=256 + embedding gather -> f16 time-major [T][B][32].
// One wave32 per (b,t). 268 MB of x streamed once: bandwidth-bound (~12 us).
// ---------------------------------------------------------------------------
__global__ __launch_bounds__(256) void k_argmax_embed(
        const float* __restrict__ x,
        const float* __restrict__ embed,
        _Float16* __restrict__ embH) {
    int lane = threadIdx.x & 31;
    int pos  = blockIdx.x * 8 + (threadIdx.x >> 5);  // pos = b*T + t, [0, B*T)
    int b    = pos >> 12;                            // / T_
    int t    = pos & (T_ - 1);

    const float* xp = x + (size_t)pos * V_;
    f32x4 u0 = *(const f32x4*)(xp + lane * 4);        // v = 4*lane .. +3
    f32x4 u1 = *(const f32x4*)(xp + 128 + lane * 4);  // v = 128+4*lane .. +3

    float bv = u0[0];
    int   bi = lane * 4;
#pragma unroll
    for (int i = 1; i < 4; ++i)
        if (u0[i] > bv) { bv = u0[i]; bi = lane * 4 + i; }
#pragma unroll
    for (int i = 0; i < 4; ++i)
        if (u1[i] > bv) { bv = u1[i]; bi = 128 + lane * 4 + i; }

    // wave reduction, first-max (lowest index) on ties
#pragma unroll
    for (int off = 16; off > 0; off >>= 1) {
        float ov = __shfl_xor(bv, off, 32);
        int   oi = __shfl_xor(bi, off, 32);
        if (ov > bv || (ov == bv && oi < bi)) { bv = ov; bi = oi; }
    }

    int r = t * B_ + b;  // time-major row
    embH[(size_t)r * H_ + lane] = (_Float16)embed[bi * H_ + lane];
}

// ---------------------------------------------------------------------------
// Kernel 2: pre = embH @ W_ih^T + (b_ih + b_hh), f32 out, time-major.
// [262144 x 32] @ [32 x 32]: one v_wmma_f32_16x16x32_f16 per 16x16 tile (K=32
// in a single instruction). 16384 waves x 2 WMMAs.
// ---------------------------------------------------------------------------
__global__ __launch_bounds__(256) void k_pregemm(
        const _Float16* __restrict__ embH,
        const float* __restrict__ W_ih,
        const float* __restrict__ b_ih,
        const float* __restrict__ b_hh,
        float* __restrict__ pre) {
    int lane = threadIdx.x & 31;
    int wave = blockIdx.x * 8 + (threadIdx.x >> 5);
    int m0   = wave * 16;
    int hiOff = (lane & 16) ? 8 : 0;

    v16h a = load_a16x32(embH + (size_t)m0 * H_, lane);

#pragma unroll
    for (int nt = 0; nt < 2; ++nt) {
        int n0 = nt * 16;
        int n  = n0 + (lane & 15);
        v16h bf = load_b_from_w32(W_ih, n0, lane);
        float bias = b_ih[n] + b_hh[n];
        v8f c;
#pragma unroll
        for (int r = 0; r < 8; ++r) c[r] = bias;
        c = __builtin_amdgcn_wmma_f32_16x16x32_f16(
                false, a, false, bf, (short)0, c, false, false);
#pragma unroll
        for (int r = 0; r < 8; ++r)
            pre[(size_t)(m0 + hiOff + r) * H_ + n] = c[r];
    }
}

// ---------------------------------------------------------------------------
// Kernel 3: sequential scan. 1 workgroup, 8 waves; h kept f16 in LDS.
// Per step: LDS A-frag -> WMMA (C = prefetched pre tile) -> relu -> LDS store,
// + per-thread 4-wide projection. pre stream pipelined 1 step in regs and
// prefetched 8 steps ahead into L2 (global_prefetch_b8).
// ---------------------------------------------------------------------------
__global__ __launch_bounds__(256) void k_rnn_scan(
        const float* __restrict__ pre,
        const float* __restrict__ h0,
        const float* __restrict__ W_hh,
        const float* __restrict__ W_p,
        const float* __restrict__ b_p,
        float* __restrict__ out) {
    __shared__ __align__(16) _Float16 hbuf[B_ * H_];  // 4 KB

    int tid   = threadIdx.x;
    int lane  = tid & 31;
    int wave  = tid >> 5;
    int m     = wave >> 1;            // M-tile 0..3
    int n0    = (wave & 1) * 16;      // N-tile 0 or 16
    int n     = n0 + (lane & 15);
    int hiOff = (lane & 16) ? 8 : 0;
    int rowBase = m * 16 + hiOff;

    // init h from input h0 (f32 -> f16)
#pragma unroll
    for (int i = 0; i < 8; ++i) {
        int idx = tid * 8 + i;
        hbuf[idx] = (_Float16)h0[idx];
    }

    // W_hh B-fragment pinned in registers for this wave's N-tile
    v16h bhh = load_b_from_w32(W_hh, n0, lane);

    // projection weights: thread owns (b = tid>>2, o = tid&3)
    int ob = tid >> 2, oo = tid & 3;
    float wp[32];
#pragma unroll
    for (int k = 0; k < 32; ++k) wp[k] = W_p[oo * 32 + k];
    float bpv = b_p[oo];

    __syncthreads();

    // pipeline: C tile for step 0
    v8f c;
#pragma unroll
    for (int r = 0; r < 8; ++r)
        c[r] = pre[(size_t)(rowBase + r) * H_ + n];

    for (int t = 0; t < T_; ++t) {
        // warm L2 eight steps ahead (256 threads x 32B covers the 8KB tile)
        if (t + 8 < T_)
            __builtin_prefetch(pre + (size_t)(t + 8) * (B_ * H_) + tid * 8, 0, 1);

        // issue next step's C loads now; waited on next iteration
        v8f cn;
        if (t + 1 < T_) {
            const float* pn = pre + (size_t)(t + 1) * (B_ * H_);
#pragma unroll
            for (int r = 0; r < 8; ++r)
                cn[r] = pn[(size_t)(rowBase + r) * H_ + n];
        }

        // h_new tile = relu(pre_t + h @ W_hh^T)
        v16h a = load_a16x32(hbuf + m * 16 * H_, lane);
        v8f d = __builtin_amdgcn_wmma_f32_16x16x32_f16(
                    false, a, false, bhh, (short)0, c, false, false);
#pragma unroll
        for (int r = 0; r < 8; ++r) d[r] = d[r] > 0.0f ? d[r] : 0.0f;

        __syncthreads();  // everyone done reading old h

#pragma unroll
        for (int r = 0; r < 8; ++r)
            hbuf[(rowBase + r) * H_ + n] = (_Float16)d[r];

        if (t == T_ - 1) {  // h_final (f32) appended after logits block
#pragma unroll
            for (int r = 0; r < 8; ++r)
                out[(size_t)T_ * B_ * OUT_ + (size_t)(rowBase + r) * H_ + n] = d[r];
        }

        __syncthreads();  // new h visible

        // logits[t*B + b][o] = h_new[b] . W_p[o] + b_p[o]
        const _Float16* hr = hbuf + ob * H_;
        V16H h01, h23;
        h01.q[0] = *(const u32x4*)(hr);
        h01.q[1] = *(const u32x4*)(hr + 8);
        h23.q[0] = *(const u32x4*)(hr + 16);
        h23.q[1] = *(const u32x4*)(hr + 24);
        float acc = bpv;
#pragma unroll
        for (int k = 0; k < 16; ++k) acc += (float)h01.h[k] * wp[k];
#pragma unroll
        for (int k = 0; k < 16; ++k) acc += (float)h23.h[k] * wp[16 + k];
        out[(size_t)t * (B_ * OUT_) + tid] = acc;  // tid == b*4 + o

        c = cn;
    }
}

// ---------------------------------------------------------------------------
extern "C" void kernel_launch(void* const* d_in, const int* in_sizes, int n_in,
                              void* d_out, int out_size, void* d_ws, size_t ws_size,
                              hipStream_t stream) {
    const float* x     = (const float*)d_in[0];
    const float* h0    = (const float*)d_in[1];
    const float* embed = (const float*)d_in[2];
    const float* W_ih  = (const float*)d_in[3];
    const float* W_hh  = (const float*)d_in[4];
    const float* b_ih  = (const float*)d_in[5];
    const float* b_hh  = (const float*)d_in[6];
    const float* W_p   = (const float*)d_in[7];
    const float* b_p   = (const float*)d_in[8];
    float* out = (float*)d_out;

    // workspace layout: embH (f16, 16 MB) | pre (f32, 32 MB)
    _Float16* embH = (_Float16*)d_ws;
    float*    pre  = (float*)((char*)d_ws + (size_t)B_ * T_ * H_ * sizeof(_Float16));

    // 262144 positions, 1 wave each, 8 waves/block
    k_argmax_embed<<<(B_ * T_) / 8, 256, 0, stream>>>(x, embed, embH);
    // 262144 rows / (8 waves * 16 rows) = 2048 blocks
    k_pregemm<<<(B_ * T_) / 128, 256, 0, stream>>>(embH, W_ih, b_ih, b_hh, pre);
    // single persistent workgroup for the sequential scan
    k_rnn_scan<<<1, 256, 0, stream>>>(pre, h0, W_hh, W_p, b_p, out);
}
